// VoxelNeXtHeadMaxPool_80659485818995
// MI455X (gfx1250) — matrix operand order repaired and checked by reading, not compiled, not run on hardware
//
#include <hip/hip_runtime.h>

// ---------------------------------------------------------------------------
// VoxelNeXt head: submanifold 3x3 conv (5 branches) + BN + ReLU + 1x1 proj +
// sigmoid/maxpool peak, on gfx1250 using V_WMMA_F32_16X16X32_F16.
// M-tile = 64 voxels/block so each B fragment feeds 4 WMMAs; branches split
// into passes of 3+2 to bound accumulator VGPRs; gather and B loads are
// software-pipelined so waits cover loads issued a full batch earlier.
// ---------------------------------------------------------------------------

#define NVOX 40000
#define CCH  128
#define HDIM 304
#define WDIM 304
#define KKER 9
#define NBR5 5
#define MT   64               // voxels per block (NVOX % MT == 0)
#define NBLK (NVOX / MT)      // 625
#define LDSROW 136            // 128 halves + 8 pad (bank-conflict avoidance)
#define NFRAG (NBR5 * KKER * 4 * 8)   // packed w1 fragments (each 512 halves)

typedef __attribute__((ext_vector_type(16))) _Float16 v16h;
typedef __attribute__((ext_vector_type(8)))  _Float16 v8h;
typedef __attribute__((ext_vector_type(8)))  float    v8f;

struct BParams {
    const float* w1;
    const float* gamma;
    const float* beta;
    const float* mean;
    const float* var;
    const float* w2;
    const float* b2;
};
struct AllParams { BParams b[NBR5]; };

__constant__ int c_oc[NBR5]  = {3, 2, 1, 3, 2};
__constant__ int c_off[NBR5] = {0, 3, 5, 6, 9};

// ---------------------------------------------------------------------------
__global__ void k_init_grid(int* __restrict__ grid) {
    int t = blockIdx.x * blockDim.x + threadIdx.x;
    if (t < HDIM * WDIM) grid[t] = -1;
}

__global__ void k_scatter(const int* __restrict__ idx, int* __restrict__ grid) {
    int n = blockIdx.x * blockDim.x + threadIdx.x;
    if (n < NVOX) {
        int y = idx[2 * n], x = idx[2 * n + 1];
        grid[y * WDIM + x] = n;
    }
}

__global__ void k_nbr(const int* __restrict__ idx, const int* __restrict__ grid,
                      int* __restrict__ nbr) {
    int t = blockIdx.x * blockDim.x + threadIdx.x;
    if (t < NVOX * KKER) {
        int n = t / KKER, k = t % KKER;
        int dy = k / 3 - 1, dx = k % 3 - 1;
        int y = idx[2 * n] + dy, x = idx[2 * n + 1] + dx;
        int v = (y >= 0 && y < HDIM && x >= 0 && x < WDIM) ? grid[y * WDIM + x] : -1;
        nbr[t] = v;
    }
}

__global__ void k_cvt(const float* __restrict__ f, _Float16* __restrict__ fh) {
    int t = blockIdx.x * blockDim.x + threadIdx.x;
    if (t < NVOX * CCH) fh[t] = (_Float16)f[t];
}

// Pack w1 (k,c,d layout) into WMMA B-fragment layout:
// fragment f = ((b*9+k)*4+ks)*8+dt holds the 32(K) x 16(N) f16 slab for
// K-chunk ks (c = ks*32..+31) and output tile dt (d = dt*16..+15).
// Within a fragment: lane L, half h  <->  c = ks*32 + 16*(L>>4) + h,
//                                         d = dt*16 + (L&15).
__global__ void k_pack(AllParams P, _Float16* __restrict__ wp) {
    int t = blockIdx.x * blockDim.x + threadIdx.x;
    if (t < NFRAG * 512) {
        int h    = t & 15;
        int lane = (t >> 4) & 31;
        int f    = t >> 9;
        int dt   = f & 7;
        int ks   = (f >> 3) & 3;
        int kb   = f >> 5;          // b*9 + k
        int k    = kb % KKER;
        int b    = kb / KKER;
        int c = ks * 32 + ((lane >> 4) << 4) + h;
        int d = dt * 16 + (lane & 15);
        wp[t] = (_Float16)P.b[b].w1[(k * CCH + c) * CCH + d];
    }
}

// ---------------------------------------------------------------------------
// One branch-group pass of the fused conv kernel.
//   NG branches starting at branch B0; 8 waves own the 8 D-tiles; each wave
//   accumulates 4 M-subtiles x NG branches of 16x16 f32.
// ---------------------------------------------------------------------------
template <int NG, int B0>
__device__ __forceinline__ void conv_pass(
    const _Float16* __restrict__ fh,
    const _Float16* __restrict__ wp,
    const int* __restrict__ nbr,
    const AllParams& P,
    float* __restrict__ out,
    float* __restrict__ hm,
    _Float16 (&gbuf)[2][MT * LDSROW],
    float* hbuf,
    int base) {
    const int tid  = threadIdx.x;
    const int lane = tid & 31;
    const int wv   = tid >> 5;            // D-tile 0..7
    const int ml   = lane & 15;
    const int hi   = lane >> 4;

    // ---- prologue gather: k = 0 into buffer 0 ----
#pragma unroll
    for (int i = 0; i < 2; ++i) {
        int u = tid + 256 * i;
        int row = u >> 3, ch = u & 7;
        int j = nbr[(base + row) * KKER + 0];
        v8h a0 = {}, a1 = {};
        if (j >= 0) {
            const v8h* s = (const v8h*)(fh + j * CCH + ch * 16);
            a0 = s[0];
            a1 = s[1];
        }
        v8h* dp = (v8h*)(&gbuf[0][0] + row * LDSROW + ch * 16);
        dp[0] = a0;
        dp[1] = a1;
    }
    __syncthreads();

    v8f acc[4][NG] = {};
    for (int k = 0; k < KKER; ++k) {
        const int cur = k & 1;
        // ---- stage gather of k+1 into registers (latency hidden by WMMAs) --
        v8h st[2][2] = {};
        int srow[2], sch[2];
        if (k + 1 < KKER) {
#pragma unroll
            for (int i = 0; i < 2; ++i) {
                int u = tid + 256 * i;
                srow[i] = u >> 3;
                sch[i]  = u & 7;
                int j = nbr[(base + srow[i]) * KKER + (k + 1)];
                if (j >= 0) {
                    const v8h* s = (const v8h*)(fh + j * CCH + sch[i] * 16);
                    st[i][0] = s[0];
                    st[i][1] = s[1];
                }
            }
        }
        // ---- compute on gbuf[cur]: 4 ks-steps, B double-buffered ----
        const _Float16* gk = &gbuf[cur][0];
        v16h Bc[NG], Bn[NG];
#pragma unroll
        for (int bi = 0; bi < NG; ++bi)
            Bc[bi] = *(const v16h*)(wp +
                (((((B0 + bi) * KKER + k) * 4 + 0) * 8 + wv) * 512) + lane * 16);
#pragma unroll
        for (int ks = 0; ks < 4; ++ks) {
            if (ks < 3) {
#pragma unroll
                for (int bi = 0; bi < NG; ++bi)
                    Bn[bi] = *(const v16h*)(wp +
                        (((((B0 + bi) * KKER + k) * 4 + (ks + 1)) * 8 + wv) * 512) +
                        lane * 16);
            }
#pragma unroll
            for (int ms = 0; ms < 4; ++ms) {
                union { v16h v; v8h h2[2]; } A;
                const _Float16* arow = gk + (ms * 16 + ml) * LDSROW;
                const int cb = ks * 32 + hi * 16;
                A.h2[0] = *(const v8h*)(arow + cb);
                A.h2[1] = *(const v8h*)(arow + cb + 8);
#pragma unroll
                for (int bi = 0; bi < NG; ++bi)
                    acc[ms][bi] = __builtin_amdgcn_wmma_f32_16x16x32_f16(
                        false, A.v, false, Bc[bi], (short)0, acc[ms][bi],
                        false, false);
            }
#pragma unroll
            for (int bi = 0; bi < NG; ++bi) Bc[bi] = Bn[bi];
        }
        // ---- commit staged gather to the other buffer ----
        if (k + 1 < KKER) {
#pragma unroll
            for (int i = 0; i < 2; ++i) {
                v8h* dp = (v8h*)(&gbuf[1 - cur][0] + srow[i] * LDSROW + sch[i] * 16);
                dp[0] = st[i][0];
                dp[1] = st[i][1];
            }
        }
        __syncthreads();
    }

    // ---- epilogue: BN + ReLU -> hbuf (aliases gather LDS) -> 1x1 proj ----
    const int d = wv * 16 + ml;
#pragma unroll
    for (int bi = 0; bi < NG; ++bi) {
        const int b = B0 + bi;
        float s  = P.b[b].gamma[d] * rsqrtf(P.b[b].var[d] + 1e-5f);
        float tt = P.b[b].beta[d] - P.b[b].mean[d] * s;
#pragma unroll
        for (int ms = 0; ms < 4; ++ms)
#pragma unroll
            for (int r = 0; r < 8; ++r) {
                int mm = ms * 16 + r + hi * 8;
                hbuf[mm * 132 + d] = fmaxf(acc[ms][bi][r] * s + tt, 0.0f);
            }
        __syncthreads();
        const int oc = c_oc[b];
        if (tid < MT * oc) {
            int mm = tid / oc, j = tid % oc;
            float sum = P.b[b].b2[j];
            const float* w2 = P.b[b].w2;
            for (int dd = 0; dd < CCH; ++dd)
                sum = fmaf(hbuf[mm * 132 + dd], w2[dd * oc + j], sum);
            int n = base + mm;
            if (b == 0) hm[n * 3 + j] = 1.0f / (1.0f + expf(-sum));
            else        out[n * 11 + c_off[b] + j] = sum;
        }
        __syncthreads();
    }
}

__launch_bounds__(256)
__global__ void k_conv(const _Float16* __restrict__ fh,
                       const _Float16* __restrict__ wp,
                       const int* __restrict__ nbr,
                       AllParams P,
                       float* __restrict__ out,
                       float* __restrict__ hm) {
    __shared__ _Float16 gbuf[2][MT * LDSROW];   // 2 x 17,408 B gather buffers
    float* hbuf = (float*)&gbuf[0][0];          // 64 x 132 f32 = 33,792 B alias
    const int base = blockIdx.x * MT;
    conv_pass<3, 0>(fh, wp, nbr, P, out, hm, gbuf, hbuf, base);
    conv_pass<2, 3>(fh, wp, nbr, P, out, hm, gbuf, hbuf, base);
}

// ---------------------------------------------------------------------------
__global__ void k_pool(const float* __restrict__ hm, const int* __restrict__ nbr,
                       float* __restrict__ out) {
    int t = blockIdx.x * blockDim.x + threadIdx.x;
    if (t < NVOX * 3) {
        int n = t / 3, j = t % 3;
        float v = hm[t];
        float p = -__builtin_inff();
        for (int k = 0; k < KKER; ++k) {
            int jn = nbr[n * KKER + k];
            if (jn >= 0) p = fmaxf(p, hm[jn * 3 + j]);
        }
        out[n * 11 + j] = (p == v) ? v : 0.0f;
    }
}

// ---------------------------------------------------------------------------
extern "C" void kernel_launch(void* const* d_in, const int* in_sizes, int n_in,
                              void* d_out, int out_size, void* d_ws, size_t ws_size,
                              hipStream_t stream) {
    (void)in_sizes; (void)n_in; (void)out_size; (void)ws_size;
    const float* features = (const float*)d_in[0];
    const int*   indices  = (const int*)d_in[1];

    AllParams P;
    for (int b = 0; b < NBR5; ++b) {
        const int base = 2 + b * 7;
        P.b[b].w1    = (const float*)d_in[base + 0];
        P.b[b].gamma = (const float*)d_in[base + 1];
        P.b[b].beta  = (const float*)d_in[base + 2];
        P.b[b].mean  = (const float*)d_in[base + 3];
        P.b[b].var   = (const float*)d_in[base + 4];
        P.b[b].w2    = (const float*)d_in[base + 5];
        P.b[b].b2    = (const float*)d_in[base + 6];
    }

    char*  ws  = (char*)d_ws;
    size_t off = 0;
    auto alloc = [&](size_t bytes) -> void* {
        void* p = ws + off;
        off = (off + bytes + 255) & ~(size_t)255;
        return p;
    };
    int*      grid = (int*)alloc((size_t)HDIM * WDIM * 4);
    int*      nbr  = (int*)alloc((size_t)NVOX * KKER * 4);
    _Float16* fh   = (_Float16*)alloc((size_t)NVOX * CCH * 2);
    _Float16* wpk  = (_Float16*)alloc((size_t)NFRAG * 512 * 2);
    float*    hm   = (float*)alloc((size_t)NVOX * 3 * 4);

    k_init_grid<<<(HDIM * WDIM + 255) / 256, 256, 0, stream>>>(grid);
    k_scatter<<<(NVOX + 255) / 256, 256, 0, stream>>>(indices, grid);
    k_nbr<<<(NVOX * KKER + 255) / 256, 256, 0, stream>>>(indices, grid, nbr);
    k_cvt<<<(NVOX * CCH + 255) / 256, 256, 0, stream>>>(features, fh);
    k_pack<<<(NFRAG * 512 + 255) / 256, 256, 0, stream>>>(P, wpk);
    k_conv<<<NBLK, 256, 0, stream>>>(fh, wpk, nbr, P, (float*)d_out, hm);
    k_pool<<<(NVOX * 3 + 255) / 256, 256, 0, stream>>>(hm, nbr, (float*)d_out);
}
